// DotProductAttention_36464272343073
// MI455X (gfx1250) — matrix operand order repaired and checked by reading, not compiled, not run on hardware
//
#include <hip/hip_runtime.h>
#include <hip/hip_bf16.h>

// ---------------------------------------------------------------------------
// Types for CDNA5 WMMA (wave32): v_wmma_f32_16x16x32_bf16
// ---------------------------------------------------------------------------
typedef __attribute__((ext_vector_type(16))) __bf16 bf16x16;
typedef __attribute__((ext_vector_type(8)))  __bf16 bf16x8;
typedef __attribute__((ext_vector_type(8)))  float  f32x8;
typedef __attribute__((ext_vector_type(4)))  unsigned int u32x4;
typedef __attribute__((ext_vector_type(4)))  int i32x4;

#if __has_builtin(__builtin_amdgcn_global_load_async_to_lds_b128)
#define HAS_ASYNC 1
#else
#define HAS_ASYNC 0
#endif

__device__ __forceinline__ unsigned short f2bf(float f) {
  unsigned u = __builtin_bit_cast(unsigned, f);
  u += 0x7fffu + ((u >> 16) & 1u);          // round-to-nearest-even
  return (unsigned short)(u >> 16);
}
__device__ __forceinline__ float bf2f(unsigned short h) {
  unsigned u = ((unsigned)h) << 16;
  return __builtin_bit_cast(float, u);
}

// 16-byte global -> LDS copy; async (ASYNCcnt) when available.
__device__ __forceinline__ void cp16(const unsigned short* g, unsigned short* l) {
#if HAS_ASYNC
  __builtin_amdgcn_global_load_async_to_lds_b128(
      (__attribute__((address_space(1))) i32x4*)g,
      (__attribute__((address_space(3))) i32x4*)l, 0, 0);
#else
  *(u32x4*)l = *(const u32x4*)g;
#endif
}
template <int CNT> __device__ __forceinline__ void wait_asynccnt() {
#if HAS_ASYNC
#if __has_builtin(__builtin_amdgcn_s_wait_asynccnt)
  __builtin_amdgcn_s_wait_asynccnt(CNT);
#else
  asm volatile("s_wait_asynccnt %0" ::"n"(CNT));
#endif
#endif
}

// A fragment (16x32 MxK, bf16): lane L holds row (L&15); 16 contiguous k at
// kb + (L>>4)*16 (matches B fragment k order)
__device__ __forceinline__ bf16x16 ld_afrag(const unsigned short* lds, int stride,
                                            int mrow, int kb, int lane) {
  const int khalf = lane >> 4;
  const unsigned short* p = lds + (size_t)(mrow + (lane & 15)) * stride + kb + khalf * 16;
  bf16x8 lo = *(const bf16x8*)p;
  bf16x8 hi = *(const bf16x8*)(p + 8);
  bf16x16 r;
#pragma unroll
  for (int i = 0; i < 8; ++i) { r[i] = lo[i]; r[i + 8] = hi[i]; }
  return r;
}
// B fragment (32x16 KxN, bf16): lane L holds col (L&15), 16 contiguous k at
// kb + (L>>4)*16. Operand stored N-major / k-contiguous.
__device__ __forceinline__ bf16x16 ld_bfrag(const unsigned short* lds, int stride,
                                            int ncol, int kb, int lane) {
  const unsigned short* p = lds + (size_t)(ncol + (lane & 15)) * stride + kb + (lane >> 4) * 16;
  return *(const bf16x16*)p;
}
__device__ __forceinline__ f32x8 wmma_bf16(bf16x16 a, bf16x16 b, f32x8 c) {
  return __builtin_amdgcn_wmma_f32_16x16x32_bf16(false, a, false, b, (short)0, c, false, false);
}

// ---------------------------------------------------------------------------
// Generic NT GEMM: C[m,n] = sum_k A[m,k] * Bw[n,k]  (A bf16, Bw bf16)
// Block tile 128x128, KC=64, 256 threads = 8 waves (2x4), wave tile 64x32.
// Double-buffered LDS, async global->LDS copies (8 x b128 per thread/tile).
// ---------------------------------------------------------------------------
template <bool BIAS, bool RELU, bool OUTBF>
__global__ __launch_bounds__(256) void gemm_nt(const unsigned short* __restrict__ A,
                                               const unsigned short* __restrict__ Bw,
                                               void* __restrict__ Cv,
                                               const float* __restrict__ bias,
                                               int M, int N, int K) {
  constexpr int BM = 128, BN = 128, KC = 64, LS = 80;  // 80 bf16 = 160B row (16B aligned)
  __shared__ __align__(16) unsigned short As[2][BM * LS];
  __shared__ __align__(16) unsigned short Bs[2][BN * LS];

  const int tid  = threadIdx.x;
  const int lane = tid & 31;
  const int wave = tid >> 5;
  const int wm = (wave >> 2) * 64;   // wave row base in tile
  const int wn = (wave & 3) * 32;    // wave col base in tile
  const int bm = blockIdx.y * BM;
  const int bn = blockIdx.x * BN;

  f32x8 acc[4][2];
#pragma unroll
  for (int i = 0; i < 4; ++i)
#pragma unroll
    for (int j = 0; j < 2; ++j)
#pragma unroll
      for (int r = 0; r < 8; ++r) acc[i][j][r] = 0.0f;

  const int lrow = tid >> 1;          // 0..127
  const int lc0  = (tid & 1) * 32;    // 0 or 32
  // Clamp OOB rows (data unused: garbage feeds only cols >= N / rows >= M,
  // which the store guard discards) so every wave issues exactly 8 asyncs.
  const int ar = (bm + lrow < M) ? (bm + lrow) : (M - 1);
  const int br = (bn + lrow < N) ? (bn + lrow) : (N - 1);

  auto stage = [&](int buf, int k0) {
    const unsigned short* ga = A + (size_t)ar * K + k0 + lc0;
    const unsigned short* gb = Bw + (size_t)br * K + k0 + lc0;
    unsigned short* la = &As[buf][lrow * LS + lc0];
    unsigned short* lb = &Bs[buf][lrow * LS + lc0];
#pragma unroll
    for (int c = 0; c < 4; ++c) cp16(ga + c * 8, la + c * 8);
#pragma unroll
    for (int c = 0; c < 4; ++c) cp16(gb + c * 8, lb + c * 8);
  };

  stage(0, 0);
  int buf = 0;
  for (int k0 = 0; k0 < K; k0 += KC) {
    if (k0 + KC < K) { stage(buf ^ 1, k0 + KC); wait_asynccnt<8>(); }
    else             { wait_asynccnt<0>(); }
    __syncthreads();

#pragma unroll
    for (int kb = 0; kb < KC; kb += 32) {
      bf16x16 af[4], bfv[2];
#pragma unroll
      for (int mi = 0; mi < 4; ++mi) af[mi] = ld_afrag(As[buf], LS, wm + mi * 16, kb, lane);
#pragma unroll
      for (int ni = 0; ni < 2; ++ni) bfv[ni] = ld_bfrag(Bs[buf], LS, wn + ni * 16, kb, lane);
#pragma unroll
      for (int mi = 0; mi < 4; ++mi)
#pragma unroll
        for (int ni = 0; ni < 2; ++ni) acc[mi][ni] = wmma_bf16(af[mi], bfv[ni], acc[mi][ni]);
    }
    __syncthreads();
    buf ^= 1;
  }

  // Epilogue: VGPR r of a 16x16 f32 tile -> row = base + (lane>>4)*8 + r, col = base + (lane&15)
#pragma unroll
  for (int mi = 0; mi < 4; ++mi)
#pragma unroll
    for (int ni = 0; ni < 2; ++ni)
#pragma unroll
      for (int r = 0; r < 8; ++r) {
        const int grow = bm + wm + mi * 16 + (lane >> 4) * 8 + r;
        const int gcol = bn + wn + ni * 16 + (lane & 15);
        if (grow < M && gcol < N) {
          float v = acc[mi][ni][r];
          if (BIAS) v += bias[gcol];
          if (RELU) v = v > 0.0f ? v : 0.0f;
          if (OUTBF) ((unsigned short*)Cv)[(size_t)grow * N + gcol] = f2bf(v);
          else       ((float*)Cv)[(size_t)grow * N + gcol] = v;
        }
      }
}

// ---------------------------------------------------------------------------
// Fused attention core per (b,h): S = QK^T*0.125 + bias ; softmax ; O = P*V
// ---------------------------------------------------------------------------
__global__ __launch_bounds__(256) void attn_fused(const unsigned short* __restrict__ qkvb,
                                                  const float* __restrict__ biasT,
                                                  unsigned short* __restrict__ Hc) {
  constexpr int LS = 80, PS = 68;
  __shared__ __align__(16) unsigned short Qs[64 * LS];  // Q, later P (bf16)
  __shared__ __align__(16) unsigned short Ks[64 * LS];
  __shared__ __align__(16) unsigned short Vt[64 * LS];  // V transposed [d][t]
  __shared__ float Ps[64 * PS];

  const int bh = blockIdx.x;
  const int b = bh >> 4, h = bh & 15;
  const int tid = threadIdx.x, lane = tid & 31, wave = tid >> 5;

  {  // stage: 4 threads per row, 16 cols each; Q/K async, V transposed by hand
    const int r = tid >> 2, c0 = (tid & 3) * 16;
    const size_t base = ((size_t)(b * 64 + r)) * 3072 + h * 192;
    cp16(qkvb + base + c0,          &Qs[r * LS + c0]);
    cp16(qkvb + base + c0 + 8,      &Qs[r * LS + c0 + 8]);
    cp16(qkvb + base + 64 + c0,     &Ks[r * LS + c0]);
    cp16(qkvb + base + 64 + c0 + 8, &Ks[r * LS + c0 + 8]);
#pragma unroll
    for (int i = 0; i < 16; ++i)
      Vt[(c0 + i) * LS + r] = qkvb[base + 128 + c0 + i];
    wait_asynccnt<0>();
  }
  __syncthreads();

  // S = Q K^T : 16 subtiles over 8 waves (2 each)
  const int s0 = wave * 2;
  f32x8 sacc[2];
#pragma unroll
  for (int s = 0; s < 2; ++s)
#pragma unroll
    for (int r = 0; r < 8; ++r) sacc[s][r] = 0.0f;
#pragma unroll
  for (int s = 0; s < 2; ++s) {
    const int mi = (s0 + s) >> 2, ni = (s0 + s) & 3;
#pragma unroll
    for (int kb = 0; kb < 64; kb += 32) {
      bf16x16 a = ld_afrag(Qs, LS, mi * 16, kb, lane);
      bf16x16 bb = ld_bfrag(Ks, LS, ni * 16, kb, lane);
      sacc[s] = wmma_bf16(a, bb, sacc[s]);
    }
  }
#pragma unroll
  for (int s = 0; s < 2; ++s) {
    const int mi = (s0 + s) >> 2, ni = (s0 + s) & 3;
#pragma unroll
    for (int r = 0; r < 8; ++r) {
      const int q = mi * 16 + (lane >> 4) * 8 + r;
      const int k = ni * 16 + (lane & 15);
      Ps[q * PS + k] = sacc[s][r] * 0.125f + biasT[(size_t)bh * 4096 + q * 64 + k];
    }
  }
  __syncthreads();

  // softmax over k (row length 64); write P as bf16 into Qs (Q dead now)
  if (tid < 64) {
    float m = -1e30f;
    for (int k = 0; k < 64; ++k) m = fmaxf(m, Ps[tid * PS + k]);
    float ssum = 0.0f;
    for (int k = 0; k < 64; ++k) { float e = __expf(Ps[tid * PS + k] - m); Ps[tid * PS + k] = e; ssum += e; }
    const float inv = 1.0f / ssum;
    for (int k = 0; k < 64; ++k) Qs[tid * LS + k] = f2bf(Ps[tid * PS + k] * inv);
  }
  __syncthreads();

  // O = P V  (B-operand columns = rows of Vt, k-contiguous)
  f32x8 oacc[2];
#pragma unroll
  for (int s = 0; s < 2; ++s)
#pragma unroll
    for (int r = 0; r < 8; ++r) oacc[s][r] = 0.0f;
#pragma unroll
  for (int s = 0; s < 2; ++s) {
    const int mi = (s0 + s) >> 2, ni = (s0 + s) & 3;
#pragma unroll
    for (int kb = 0; kb < 64; kb += 32) {
      bf16x16 a = ld_afrag(Qs, LS, mi * 16, kb, lane);
      bf16x16 bb = ld_bfrag(Vt, LS, ni * 16, kb, lane);
      oacc[s] = wmma_bf16(a, bb, oacc[s]);
    }
  }
#pragma unroll
  for (int s = 0; s < 2; ++s) {
    const int mi = (s0 + s) >> 2, ni = (s0 + s) & 3;
#pragma unroll
    for (int r = 0; r < 8; ++r) {
      const int q = mi * 16 + (lane >> 4) * 8 + r;
      const int d = ni * 16 + (lane & 15);
      Hc[((size_t)(b * 64 + q)) * 1024 + h * 64 + d] = f2bf(oacc[s][r]);
    }
  }
}

// ---------------------------------------------------------------------------
// Small helper kernels
// ---------------------------------------------------------------------------
__global__ void k_convert_bf(const float* __restrict__ src, unsigned short* __restrict__ dst, int n) {
  int i = blockIdx.x * blockDim.x + threadIdx.x;
  if (i < n) dst[i] = f2bf(src[i]);
}

// Wb[e, 0:256]=prep_w; [256:768]=0.125*fq_w; [768:1280]=0.125*fk_w  (bf16)
__global__ void k_build_wb(const float* __restrict__ prep_w, const float* __restrict__ fq_w,
                           const float* __restrict__ fk_w, unsigned short* __restrict__ wb) {
  int i = blockIdx.x * blockDim.x + threadIdx.x;
  if (i >= 4096 * 1280) return;
  const int r = i / 1280, c = i % 1280;
  float v;
  if (c < 256)      v = prep_w[r * 256 + c];
  else if (c < 768) v = 0.125f * fq_w[r * 512 + (c - 256)];
  else              v = 0.125f * fk_w[r * 512 + (c - 768)];
  wb[i] = f2bf(v);
}
__global__ void k_build_biasc(const float* __restrict__ prep_b, const float* __restrict__ fq_b,
                              const float* __restrict__ fk_b, float* __restrict__ bc) {
  int i = blockIdx.x * blockDim.x + threadIdx.x;
  if (i < 4096) bc[i] = prep_b[i] + 0.125f * (fq_b[i] + fk_b[i]);
}

// LayerNorm, one block (256 thr) per row; bf16 output.
// mode 1 scatters into abuf layout [(b,h)][0:256].
__global__ __launch_bounds__(256) void k_layernorm(const float* __restrict__ in,
                                                   unsigned short* __restrict__ out,
                                                   const float* __restrict__ g,
                                                   const float* __restrict__ be,
                                                   int cols, int mode) {
  __shared__ float red[256];
  const int row = blockIdx.x, tid = threadIdx.x;
  const float* x = in + (size_t)row * cols;
  float s = 0.0f, s2 = 0.0f;
  for (int c = tid; c < cols; c += 256) { float v = x[c]; s += v; s2 += v * v; }
  red[tid] = s; __syncthreads();
  for (int o = 128; o > 0; o >>= 1) { if (tid < o) red[tid] += red[tid + o]; __syncthreads(); }
  const float mean = red[0] / cols; __syncthreads();
  red[tid] = s2; __syncthreads();
  for (int o = 128; o > 0; o >>= 1) { if (tid < o) red[tid] += red[tid + o]; __syncthreads(); }
  const float var = red[0] / cols - mean * mean;
  const float inv = rsqrtf(var + 1e-5f);
  for (int c = tid; c < cols; c += 256) {
    const float v = (x[c] - mean) * inv * g[c] + be[c];
    if (mode == 0) out[(size_t)row * cols + c] = f2bf(v);
    else out[((size_t)row * 16 + (c >> 8)) * 1280 + (c & 255)] = f2bf(v);
  }
}

// Qsum[b,h,d] = sum_t Q ; Ksum likewise (from bf16 qkv)
__global__ void k_qksum(const unsigned short* __restrict__ qkvb,
                        float* __restrict__ Qsum, float* __restrict__ Ksum) {
  int i = blockIdx.x * blockDim.x + threadIdx.x;
  if (i >= 16384 * 64) return;
  const int d = i & 63, bh = i >> 6, b = bh >> 4, hh = bh & 15;
  const size_t base = ((size_t)b * 64) * 3072 + hh * 192 + d;
  float sq = 0.0f, sk = 0.0f;
  for (int t = 0; t < 64; ++t) {
    sq += bf2f(qkvb[base + (size_t)t * 3072]);
    sk += bf2f(qkvb[base + (size_t)t * 3072 + 64]);
  }
  Qsum[i] = sq; Ksum[i] = sk;
}

// rqz/rkz[b,h,z] -> abuf slices [256:768),[768:1280)  (bf16)
__global__ void k_rz(const float* __restrict__ Qsum, const float* __restrict__ Ksum,
                     const float* __restrict__ qp, const float* __restrict__ kp,
                     unsigned short* __restrict__ abuf) {
  int i = blockIdx.x * blockDim.x + threadIdx.x;
  if (i >= 16384 * 512) return;
  const int z = i & 511, bh = i >> 9, hh = bh & 15;
  float aq = 0.0f, ak = 0.0f;
  for (int d = 0; d < 64; ++d) {
    aq += Qsum[bh * 64 + d] * qp[((size_t)d * 16 + hh) * 512 + z];
    ak += Ksum[bh * 64 + d] * kp[((size_t)d * 16 + hh) * 512 + z];
  }
  abuf[(size_t)bh * 1280 + 256 + z] = f2bf(aq);
  abuf[(size_t)bh * 1280 + 768 + z] = f2bf(ak);
}

// ---------------------------------------------------------------------------
extern "C" void kernel_launch(void* const* d_in, const int* in_sizes, int n_in,
                              void* d_out, int out_size, void* d_ws, size_t ws_size,
                              hipStream_t stream) {
  (void)in_sizes; (void)n_in; (void)out_size; (void)ws_size;
  const float* x        = (const float*)d_in[0];
  const float* query    = (const float*)d_in[1];
  const float* qkv_w    = (const float*)d_in[5];
  const float* wh_w     = (const float*)d_in[6];
  const float* wh_b     = (const float*)d_in[7];
  const float* sm1_w    = (const float*)d_in[8];
  const float* sm1_b    = (const float*)d_in[9];
  const float* sm2_w    = (const float*)d_in[10];
  const float* sm2_b    = (const float*)d_in[11];
  const float* ln1_g    = (const float*)d_in[12];
  const float* ln1_b    = (const float*)d_in[13];
  const float* sm3_w    = (const float*)d_in[14];
  const float* sm3_b    = (const float*)d_in[15];
  const float* ln2_g    = (const float*)d_in[16];
  const float* ln2_b    = (const float*)d_in[17];
  const float* prep_w   = (const float*)d_in[18];
  const float* prep_b   = (const float*)d_in[19];
  const float* rpe_q_p  = (const float*)d_in[20];
  const float* rpe_k_p  = (const float*)d_in[21];
  const float* rpe_fq_w = (const float*)d_in[22];
  const float* rpe_fq_b = (const float*)d_in[23];
  const float* rpe_fk_w = (const float*)d_in[24];
  const float* rpe_fk_b = (const float*)d_in[25];

  char* ws = (char*)d_ws;
  size_t off = 0;
  auto alloc = [&](size_t bytes) -> char* {
    char* p = ws + off; off += (bytes + 255) & ~(size_t)255; return p;
  };
  unsigned short* qkvb     = (unsigned short*)alloc((size_t)65536 * 3072 * 2);  // bf16 QKV
  float*          biasT    = (float*)alloc((size_t)16384 * 4096 * 4);           // sg + scaled rpe
  unsigned short* query_bf = (unsigned short*)alloc((size_t)65536 * 1024 * 2);
  unsigned short* x_bf     = (unsigned short*)alloc((size_t)65536 * 1024 * 2);
  unsigned short* Hc       = (unsigned short*)alloc((size_t)65536 * 1024 * 2);
  unsigned short* abuf     = (unsigned short*)alloc((size_t)16384 * 1280 * 2);  // [s|rqz|rkz]
  unsigned short* s1       = (unsigned short*)alloc((size_t)65536 * 32 * 2);
  float*          s2       = (float*)alloc((size_t)1024 * 256 * 4);
  unsigned short* s2n      = (unsigned short*)alloc((size_t)1024 * 256 * 2);
  float*          s3       = (float*)alloc((size_t)1024 * 4096 * 4);
  float*          Qsum     = (float*)alloc((size_t)16384 * 64 * 4);
  float*          Ksum     = (float*)alloc((size_t)16384 * 64 * 4);
  unsigned short* qkvw_bf  = (unsigned short*)alloc((size_t)3072 * 1024 * 2);
  unsigned short* whw_bf   = (unsigned short*)alloc((size_t)1024 * 1024 * 2);
  unsigned short* sm1w_bf  = (unsigned short*)alloc((size_t)32 * 1024 * 2);
  unsigned short* sm2w_bf  = (unsigned short*)alloc((size_t)256 * 2048 * 2);
  unsigned short* sm3w_bf  = (unsigned short*)alloc((size_t)4096 * 256 * 2);
  unsigned short* wb_bf    = (unsigned short*)alloc((size_t)4096 * 1280 * 2);
  float*          biasc    = (float*)alloc(4096 * 4);

  auto cdiv = [](int a, int b) { return (a + b - 1) / b; };

  // activations + weights -> bf16
  k_convert_bf<<<cdiv(65536 * 1024, 256), 256, 0, stream>>>(query, query_bf, 65536 * 1024);
  k_convert_bf<<<cdiv(65536 * 1024, 256), 256, 0, stream>>>(x, x_bf, 65536 * 1024);
  k_convert_bf<<<cdiv(3072 * 1024, 256), 256, 0, stream>>>(qkv_w, qkvw_bf, 3072 * 1024);
  k_convert_bf<<<cdiv(1024 * 1024, 256), 256, 0, stream>>>(wh_w, whw_bf, 1024 * 1024);
  k_convert_bf<<<cdiv(32 * 1024, 256), 256, 0, stream>>>(sm1_w, sm1w_bf, 32 * 1024);
  k_convert_bf<<<cdiv(256 * 2048, 256), 256, 0, stream>>>(sm2_w, sm2w_bf, 256 * 2048);
  k_convert_bf<<<cdiv(4096 * 256, 256), 256, 0, stream>>>(sm3_w, sm3w_bf, 4096 * 256);
  k_build_wb<<<cdiv(4096 * 1280, 256), 256, 0, stream>>>(prep_w, rpe_fq_w, rpe_fk_w, wb_bf);
  k_build_biasc<<<cdiv(4096, 256), 256, 0, stream>>>(prep_b, rpe_fq_b, rpe_fk_b, biasc);

  // QKV projection: [65536,1024] x [3072,1024]^T -> bf16
  gemm_nt<false, false, true><<<dim3(3072 / 128, 65536 / 128), 256, 0, stream>>>(
      query_bf, qkvw_bf, qkvb, nullptr, 65536, 3072, 1024);
  // smolgen chain
  gemm_nt<true, false, true><<<dim3(1, 65536 / 128), 256, 0, stream>>>(
      x_bf, sm1w_bf, s1, sm1_b, 65536, 32, 1024);
  gemm_nt<true, true, false><<<dim3(2, 1024 / 128), 256, 0, stream>>>(
      s1, sm2w_bf, s2, sm2_b, 1024, 256, 2048);
  k_layernorm<<<1024, 256, 0, stream>>>(s2, s2n, ln1_g, ln1_b, 256, 0);
  gemm_nt<true, true, false><<<dim3(4096 / 128, 1024 / 128), 256, 0, stream>>>(
      s2n, sm3w_bf, s3, sm3_b, 1024, 4096, 256);
  k_layernorm<<<1024, 256, 0, stream>>>(s3, abuf, ln2_g, ln2_b, 4096, 1);
  // RPE reductions into abuf
  k_qksum<<<cdiv(16384 * 64, 256), 256, 0, stream>>>(qkvb, Qsum, Ksum);
  k_rz<<<cdiv(16384 * 512, 256), 256, 0, stream>>>(Qsum, Ksum, rpe_q_p, rpe_k_p, abuf);
  // combined bias GEMM: [16384,1280] x [4096,1280]^T
  gemm_nt<true, false, false><<<dim3(4096 / 128, 16384 / 128), 256, 0, stream>>>(
      abuf, wb_bf, biasT, biasc, 16384, 4096, 1280);
  // fused attention core, one WG per (b,h)
  attn_fused<<<16384, 256, 0, stream>>>(qkvb, biasT, Hc);
  // output projection
  gemm_nt<true, false, false><<<dim3(1024 / 128, 65536 / 128), 256, 0, stream>>>(
      Hc, whw_bf, d_out, wh_b, 65536, 1024, 1024);
}
